// FcGagaLayer_37306085933762
// MI455X (gfx1250) — matrix operand order repaired and checked by reading, not compiled, not run on hardware
//
// FC-GAGA layer for MI455X (gfx1250): bf16 WMMA GEMMs + TDM tensor_load_to_lds
// staging, wave32, double-buffered LDS pipeline, batched B-fragment loads with
// sched_barrier-enforced load/WMMA separation.
#include <hip/hip_runtime.h>
#include <cstdint>
#include <cstddef>

#define B_    128
#define N_    207
#define W_    12
#define H_    12
#define D_    64
#define HID_  128
#define IN_SZ 2560            // W + N*W + D
#define ROWS  (B_ * N_)       // 26496 = 207 * 128 (exact multiple of 128)
#define EPS_  10.0f

#define LDS_ROW   272         // 128 bf16 = 256B + 16B pad (bank-conflict break)
#define TILE_BYTES (128 * LDS_ROW)   // 34816

typedef __attribute__((ext_vector_type(16))) __bf16  v16bf;
typedef __attribute__((ext_vector_type(8)))  float   v8f;
typedef __attribute__((ext_vector_type(4)))  unsigned int uint4v;
typedef __attribute__((ext_vector_type(4)))  int     int4v;
typedef __attribute__((ext_vector_type(8)))  int     int8v;

#ifndef __has_builtin
#define __has_builtin(x) 0
#endif
#if __has_builtin(__builtin_amdgcn_tensor_load_to_lds) && __has_builtin(__builtin_amdgcn_s_wait_tensorcnt)
#define HAVE_TDM 1
#else
#define HAVE_TDM 0
#endif

#if __has_builtin(__builtin_amdgcn_sched_barrier)
#define SCHED_FENCE() __builtin_amdgcn_sched_barrier(0)
#else
#define SCHED_FENCE()
#endif

// ---------------------------------------------------------------------------
// TDM: 2D tile load global->LDS, bf16 elements, 128-wide rows, LDS pad
// +4 dwords per 64 dwords (row stride 272B). D# per CDNA5 ISA §8.3/8.4.
// This toolchain (clang-23 / therock-10.0) uses the 6-arg builtin:
//   (uint32x4 g0, int32x8 g1, int32x4 g2, int32x4 g3, int32x8 g4, i32 cpol)
// g2/g3/g4 are zero-filled: tensors here are 2D, no gather/iterate.
// ---------------------------------------------------------------------------
#if HAVE_TDM
__device__ __forceinline__ void tdm_load_tile_bf16(uint32_t lds_byte_addr,
                                                   const void* gptr,
                                                   uint32_t stride_elems) {
  const uint32_t tile_w = 128, tile_h = 128;
  uint64_t ga = (uint64_t)(uintptr_t)gptr;
  uint4v g0;
  g0.x = 1u;                                   // count=1 (valid descriptor)
  g0.y = lds_byte_addr;                        // lds_addr [63:32]
  g0.z = (uint32_t)ga;                         // global_addr low
  g0.w = (uint32_t)(ga >> 32) | (2u << 30);    // global_addr hi | type=2
  int8v g1;
  // data_size=1 (2B) | pad_enable | pad_interval=64dw (code5) | pad_amount=4dw (code3)
  g1.x = (int)((1u << 16) | (1u << 20) | (5u << 22) | (3u << 25));
  g1.y = (int)((tile_w & 0xffffu) << 16);                  // tensor_dim0 lo16
  g1.z = (int)(((tile_h & 0xffffu) << 16) | (tile_w >> 16)); // td0 hi | td1 lo
  g1.w = (int)(((tile_w & 0xffffu) << 16) | (tile_h >> 16)); // td1 hi | tile_dim0
  g1[4] = (int)(tile_h & 0xffffu);                         // tile_dim1 | tile_dim2=0
  g1[5] = (int)stride_elems;                               // tensor_dim0_stride lo32
  g1[6] = 0;                                               // stride hi | td1_stride lo
  g1[7] = 0;
  int4v g2 = {0, 0, 0, 0};
  int4v g3 = {0, 0, 0, 0};
  int8v g4;
  g4.x = 0; g4.y = 0; g4.z = 0; g4.w = 0;
  g4[4] = 0; g4[5] = 0; g4[6] = 0; g4[7] = 0;
  __builtin_amdgcn_tensor_load_to_lds(g0, g1, g2, g3, g4, 0);
}
#endif

// Stage a 128x128 bf16 tile (row stride `stride_elems` in global) into LDS
// (row stride 272B). TDM path issues one DMA from wave 0 / lane 0.
__device__ __forceinline__ void stage(char* lds_dst, const __bf16* g,
                                      uint32_t stride_elems) {
#if HAVE_TDM
  if (threadIdx.x == 0)
    tdm_load_tile_bf16((uint32_t)(uintptr_t)lds_dst, (const void*)g, stride_elems);
#else
  for (uint32_t i = threadIdx.x; i < 128u * 16u; i += blockDim.x) {
    uint32_t r = i >> 4, c = (i & 15u) << 3;   // 8 bf16 = 16B chunks
    *(uint4v*)(lds_dst + r * LDS_ROW + c * 2) =
        *(const uint4v*)(g + (size_t)r * stride_elems + c);
  }
#endif
}

__device__ __forceinline__ void stage_wait() {
#if HAVE_TDM
  if (threadIdx.x == 0) __builtin_amdgcn_s_wait_tensorcnt(0);
#endif
  __syncthreads();
}

// ---------------------------------------------------------------------------
// One 128-K chunk of a GEMM strip: wave computes 16 rows x 128 cols.
// A: LDS, row-major M x K (rows = 272B). B: LDS, N-major (row n = K values).
// Fragment layouts per CDNA5 ISA §7.12.2 (16-bit A 16x32, B 32x16, wave32).
// All 8 B fragments of a K-step are loaded as one batch (16 ds_load_b128)
// with a sched_barrier before the 8 back-to-back WMMAs: the scheduler cannot
// sink loads past the barrier, so all fragments stay simultaneously live in
// distinct registers and the LDS loads drain behind the matrix pipe.
// ---------------------------------------------------------------------------
union Frag { v16bf bf; uint4v u[2]; };

__device__ __forceinline__ void gemm_chunk(v8f* acc, const char* A,
                                           const char* Bt, int m0, int mlo,
                                           int hh) {
#pragma unroll
  for (int ks = 0; ks < 4; ++ks) {
    Frag af;
    const char* arow = A + (m0 + mlo) * LDS_ROW + (ks * 32 + hh * 8) * 2;
    af.u[0] = *(const uint4v*)(arow);        // K = kb .. kb+7
    af.u[1] = *(const uint4v*)(arow + 32);   // K = kb+16 .. kb+23
    const char* bbase = Bt + mlo * LDS_ROW + (ks * 32 + hh * 8) * 2;
    Frag bf[8];
#pragma unroll
    for (int t = 0; t < 8; ++t) {
      const char* brow = bbase + t * 16 * LDS_ROW;
      bf[t].u[0] = *(const uint4v*)(brow);
      bf[t].u[1] = *(const uint4v*)(brow + 32);
    }
    SCHED_FENCE();   // keep all 18 ds_load_b128 ahead of the WMMA burst
#pragma unroll
    for (int t = 0; t < 8; ++t) {
      acc[t] = __builtin_amdgcn_wmma_f32_16x16x32_bf16(
          false, af.bf, false, bf[t].bf, (short)0, acc[t], false, false);
    }
    SCHED_FENCE();
  }
}

// ---------------------------------------------------------------------------
// Fused N-BEATS block: h = relu^3(MLP(X)), F += h@Wf+bf, BC = relu(X - h@Wb-bb)
// One workgroup = 128 rows. 8 waves x (16 rows x 128 cols) per GEMM.
// LDS map (dynamic): [0,1]=A ping/pong, [2,3]=B ping/pong, [4]=H.
// ---------------------------------------------------------------------------
struct BlockArgs {
  const __bf16* Xb;  const float* Xf;
  const __bf16 *W0t, *W1t, *W2t, *WbT;   // pre-transposed bf16 (N-major)
  const float *b0, *b1, *b2;
  const float *Wf, *bfv;                 // forecast (128x12), bias(12)
  const float *bb;                       // backcast bias (2560)
  float* Facc; int addF;
  float* BCf; __bf16* BCb; int writeBCb;
};

__global__ __launch_bounds__(256) void fc_block_kernel(BlockArgs a) {
  extern __shared__ char smem[];
  // NOTE: no arrays of LDS pointers (lld cannot relocate addrspacecast
  // initializers); buffers addressed as smem + idx*TILE_BYTES.
  char* Hb = smem + 4 * (size_t)TILE_BYTES;

  const int tid = threadIdx.x, lane = tid & 31, wave = tid >> 5;
  const int row0 = blockIdx.x * 128;
  const int m0 = wave * 16, mlo = lane & 15, hh = lane >> 4;

  v8f acc[8];
#pragma unroll
  for (int t = 0; t < 8; ++t)
#pragma unroll
    for (int e = 0; e < 8; ++e) acc[t][e] = 0.f;

  // ---------------- Layer 0: H = relu(X @ W0 + b0), K = 2560 ----------------
  stage(smem + 0 * TILE_BYTES, a.Xb + (size_t)row0 * IN_SZ, IN_SZ);
  stage(smem + 2 * TILE_BYTES, a.W0t, IN_SZ);
  for (int kc = 0; kc < 20; ++kc) {
    const int cur = kc & 1, nxt = cur ^ 1;
    stage_wait();
    if (kc + 1 < 20) {
      stage(smem + (size_t)nxt * TILE_BYTES,
            a.Xb + (size_t)row0 * IN_SZ + (kc + 1) * 128, IN_SZ);
      stage(smem + (size_t)(2 + nxt) * TILE_BYTES,
            a.W0t + (size_t)(kc + 1) * 128, IN_SZ);
    }
    gemm_chunk(acc, smem + (size_t)cur * TILE_BYTES,
               smem + (size_t)(2 + cur) * TILE_BYTES, m0, mlo, hh);
    __syncthreads();
  }
#pragma unroll
  for (int t = 0; t < 8; ++t) {
    float bias = a.b0[t * 16 + mlo];
#pragma unroll
    for (int r = 0; r < 8; ++r) {
      float v = acc[t][r] + bias;
      v = v > 0.f ? v : 0.f;
      *(__bf16*)(Hb + (m0 + hh * 8 + r) * LDS_ROW + (t * 16 + mlo) * 2) = (__bf16)v;
    }
  }
  __syncthreads();

  // ---------------- Layers 1 & 2: 128x128 ----------------
  stage(smem + 2 * TILE_BYTES, a.W1t, 128);
  stage_wait();
  stage(smem + 3 * TILE_BYTES, a.W2t, 128);  // prefetch W2 during layer 1
  for (int l = 0; l < 2; ++l) {
    const float* lbias = (l == 0) ? a.b1 : a.b2;
#pragma unroll
    for (int t = 0; t < 8; ++t)
#pragma unroll
      for (int e = 0; e < 8; ++e) acc[t][e] = 0.f;
    gemm_chunk(acc, Hb, smem + (size_t)(2 + l) * TILE_BYTES, m0, mlo, hh);
    __syncthreads();
#pragma unroll
    for (int t = 0; t < 8; ++t) {
      float bias = lbias[t * 16 + mlo];
#pragma unroll
      for (int r = 0; r < 8; ++r) {
        float v = acc[t][r] + bias;
        v = v > 0.f ? v : 0.f;
        *(__bf16*)(Hb + (m0 + hh * 8 + r) * LDS_ROW + (t * 16 + mlo) * 2) = (__bf16)v;
      }
    }
    __syncthreads();
    if (l == 0) stage_wait();  // W2 tile ready
  }

  // ---------------- Backcast GEMM prefetch + forecast (VALU) ----------------
  stage(smem + 2 * TILE_BYTES, a.WbT, 128);  // overlap DMA with forecast
  {
    int m = tid >> 1, c0 = (tid & 1) * 6;
    const __bf16* hrow = (const __bf16*)(Hb + m * LDS_ROW);
    float f[6];
#pragma unroll
    for (int c = 0; c < 6; ++c) f[c] = a.bfv[c0 + c];
    for (int k = 0; k < 128; ++k) {
      float hv = (float)hrow[k];
#pragma unroll
      for (int c = 0; c < 6; ++c) f[c] += hv * a.Wf[k * 12 + c0 + c];
    }
    size_t base = (size_t)(row0 + m) * 12 + c0;
    if (a.addF) {
#pragma unroll
      for (int c = 0; c < 6; ++c) a.Facc[base + c] += f[c];
    } else {
#pragma unroll
      for (int c = 0; c < 6; ++c) a.Facc[base + c] = f[c];
    }
  }

  // ---------------- Backcast: BC = relu(X - (H @ Wb + bb)), N = 2560 --------
  for (int nc = 0; nc < 20; ++nc) {
    const int cur = nc & 1, nxt = cur ^ 1;
    stage_wait();
    if (nc + 1 < 20)
      stage(smem + (size_t)(2 + nxt) * TILE_BYTES,
            a.WbT + (size_t)(nc + 1) * 128 * 128, 128);
    v8f c8[8];
#pragma unroll
    for (int t = 0; t < 8; ++t)
#pragma unroll
      for (int e = 0; e < 8; ++e) c8[t][e] = 0.f;
    gemm_chunk(c8, Hb, smem + (size_t)(2 + cur) * TILE_BYTES, m0, mlo, hh);
#pragma unroll
    for (int t = 0; t < 8; ++t) {
      int col = nc * 128 + t * 16 + mlo;
      float bias = a.bb[col];
#pragma unroll
      for (int r = 0; r < 8; ++r) {
        size_t gi = (size_t)(row0 + m0 + hh * 8 + r) * IN_SZ + col;
        float v = a.Xf[gi] - (c8[t][r] + bias);
        v = v > 0.f ? v : 0.f;
        a.BCf[gi] = v;
        if (a.writeBCb) a.BCb[gi] = (__bf16)v;
      }
    }
    __syncthreads();
  }
}

// ---------------------------------------------------------------------------
// Prologue kernels
// ---------------------------------------------------------------------------
__global__ void fc_transpose_bf16(const float* src, __bf16* dst, int R, int C) {
  size_t total = (size_t)R * C;
  for (size_t i = (size_t)blockIdx.x * 256 + threadIdx.x; i < total;
       i += (size_t)gridDim.x * 256) {
    int r = (int)(i / C), c = (int)(i % C);
    dst[(size_t)c * R + r] = (__bf16)src[i];
  }
}

__global__ __launch_bounds__(256) void fc_dp_kernel(const float* emb, float* dp) {
  __shared__ float e0[D_];
  int n = blockIdx.x;
  if (threadIdx.x < D_) e0[threadIdx.x] = emb[n * D_ + threadIdx.x];
  __syncthreads();
  int m = threadIdx.x;
  if (m < N_) {
    float s = 0.f;
#pragma unroll 8
    for (int i = 0; i < D_; ++i) s += e0[i] * emb[m * D_ + i];
    dp[n * N_ + m] = expf(EPS_ * s);
  }
}

__global__ __launch_bounds__(128) void fc_tg_kernel(
    const float* hist_in, const int* node_id, const float* tod,
    const float* emb, const float* Wt1, const float* bt1, const float* Wt2,
    const float* bt2, const float* Wt3, const float* bt3, float* histO,
    float* levelO, float* tgfwdO, float* Xf, __bf16* Xb) {
  __shared__ float in[76];
  __shared__ float h[HID_];
  __shared__ float hs[W_];
  __shared__ float lvl_s;
  int row = blockIdx.x;
  int t = threadIdx.x;
  int nid = node_id[row];
  if (t < D_) in[t] = emb[nid * D_ + t];
  else if (t < 76) in[t] = tod[(size_t)row * W_ + (t - D_)];
  __syncthreads();
  float s = bt1[t];
  for (int i = 0; i < 76; ++i) s += in[i] * Wt1[i * HID_ + t];
  h[t] = s > 0.f ? s : 0.f;
  if (t < D_) {  // emb tail of X: cols 2496..2559
    float e = in[t];
    Xf[(size_t)row * IN_SZ + (W_ + N_ * W_) + t] = e;
    Xb[(size_t)row * IN_SZ + (W_ + N_ * W_) + t] = (__bf16)e;
  }
  __syncthreads();
  if (t < W_) {
    float fw = bt2[t], bw = bt3[t];
    for (int k = 0; k < HID_; ++k) {
      float hv = h[k];
      fw += hv * Wt2[k * H_ + t];
      bw += hv * Wt3[k * W_ + t];
    }
    tgfwdO[(size_t)row * H_ + t] = fw;
    float hi = hist_in[(size_t)row * W_ + t] / (1.f + bw);
    histO[(size_t)row * W_ + t] = hi;
    hs[t] = hi;
  }
  __syncthreads();
  if (t == 0) {
    float lv = hs[0];
#pragma unroll
    for (int k = 1; k < W_; ++k) lv = fmaxf(lv, hs[k]);
    levelO[row] = lv;
    lvl_s = lv;
  }
  __syncthreads();
  if (t < W_) {
    float hn = hs[t] / lvl_s;
    if (__builtin_isinf(hn)) hn = 0.f;
    Xf[(size_t)row * IN_SZ + t] = hn;
    Xb[(size_t)row * IN_SZ + t] = (__bf16)hn;
  }
}

__global__ __launch_bounds__(256) void fc_anh_kernel(const float* histO,
                                                     const float* levelO,
                                                     const float* dp, float* Xf,
                                                     __bf16* Xb) {
  __shared__ float dprow[N_];
  int row = blockIdx.x, b = row / N_, n = row % N_;
  for (int j = threadIdx.x; j < N_; j += 256) dprow[j] = dp[n * N_ + j];
  float lvl = levelO[row];
  __syncthreads();
  const float* hb = histO + (size_t)b * (N_ * W_);
  size_t xbase = (size_t)row * IN_SZ + W_;
  for (int k = threadIdx.x; k < N_ * W_; k += 256) {
    __builtin_prefetch(&hb[k + 1024], 0, 1);   // -> global_prefetch_b8
    float v = hb[k] * dprow[k / W_];           // hist[b,j,w] * dp[n,j]
    float av = (v - lvl) / lvl;
    if (__builtin_isinf(av)) av = 0.f;         // jnp.where(isinf, 0, .)
    av = av > 0.f ? av : 0.f;                  // relu
    Xf[xbase + k] = av;
    Xb[xbase + k] = (__bf16)av;
  }
}

__global__ void fc_forecast_kernel(const float* F, const float* tgfwd,
                                   const float* level, float* out) {
  int i = blockIdx.x * 256 + threadIdx.x;
  if (i < ROWS * H_)
    out[i] = (1.f + tgfwd[i]) * (F[i] * level[i / H_]);
}

// ---------------------------------------------------------------------------
// Launch. d_in order (dict insertion order; nested dicts as sorted pytrees):
//  0 history_in, 1 node_id_in, 2 time_of_day_in,
//  per block i (base=3+10i): bcW,bcB,fW,fB,l0W,l0B,l1W,l1B,l2W,l2B,
//  33 emb, 34 tg1W, 35 tg1b, 36 tg2W, 37 tg2b, 38 tg3W, 39 tg3b.
// d_out: backcast (ROWS*2560) then forecast (ROWS*12), f32.
// ---------------------------------------------------------------------------
extern "C" void kernel_launch(void* const* d_in, const int* in_sizes, int n_in,
                              void* d_out, int out_size, void* d_ws,
                              size_t ws_size, hipStream_t stream) {
  const float* hist_in = (const float*)d_in[0];
  const int* node_id = (const int*)d_in[1];
  const float* tod = (const float*)d_in[2];
  const float* emb = (const float*)d_in[33];
  const float *tg1W = (const float*)d_in[34], *tg1b = (const float*)d_in[35];
  const float *tg2W = (const float*)d_in[36], *tg2b = (const float*)d_in[37];
  const float *tg3W = (const float*)d_in[38], *tg3b = (const float*)d_in[39];

  char* p = (char*)d_ws;
  auto carve = [&](size_t bytes) {
    char* r = p;
    p += (bytes + 255) & ~(size_t)255;
    return r;
  };
  const size_t XE = (size_t)ROWS * IN_SZ;
  float*  xAf = (float*)carve(XE * 4);
  float*  xBf = (float*)carve(XE * 4);
  __bf16* xAb = (__bf16*)carve(XE * 2);
  __bf16* xBb = (__bf16*)carve(XE * 2);
  float*  histO = (float*)carve((size_t)ROWS * W_ * 4);
  float*  levelO = (float*)carve((size_t)ROWS * 4);
  float*  tgfwdO = (float*)carve((size_t)ROWS * H_ * 4);
  float*  Facc = (float*)carve((size_t)ROWS * H_ * 4);
  float*  dp = (float*)carve((size_t)N_ * N_ * 4);
  __bf16* W0t[3]; __bf16* W1t[3]; __bf16* W2t[3]; __bf16* WbT[3];
  for (int i = 0; i < 3; ++i) {
    W0t[i] = (__bf16*)carve((size_t)IN_SZ * HID_ * 2);
    W1t[i] = (__bf16*)carve((size_t)HID_ * HID_ * 2);
    W2t[i] = (__bf16*)carve((size_t)HID_ * HID_ * 2);
    WbT[i] = (__bf16*)carve((size_t)IN_SZ * HID_ * 2);
  }

  // Weight transpose + bf16 convert (B operands stored N-major)
  for (int i = 0; i < 3; ++i) {
    int base = 3 + 10 * i;
    fc_transpose_bf16<<<512, 256, 0, stream>>>((const float*)d_in[base + 4], W0t[i], IN_SZ, HID_);
    fc_transpose_bf16<<<64, 256, 0, stream>>>((const float*)d_in[base + 6], W1t[i], HID_, HID_);
    fc_transpose_bf16<<<64, 256, 0, stream>>>((const float*)d_in[base + 8], W2t[i], HID_, HID_);
    fc_transpose_bf16<<<512, 256, 0, stream>>>((const float*)d_in[base + 0], WbT[i], HID_, IN_SZ);
  }

  fc_dp_kernel<<<N_, 256, 0, stream>>>(emb, dp);
  fc_tg_kernel<<<ROWS, 128, 0, stream>>>(hist_in, node_id, tod, emb, tg1W, tg1b,
                                         tg2W, tg2b, tg3W, tg3b, histO, levelO,
                                         tgfwdO, xAf, xAb);
  fc_anh_kernel<<<ROWS, 256, 0, stream>>>(histO, levelO, dp, xAf, xAb);

  float* out_bc = (float*)d_out;                 // ROWS * IN_SZ
  float* out_fc = out_bc + XE;                   // ROWS * H_
  const size_t SMEM = 5 * TILE_BYTES;            // 2xA + 2xB + H = 174080 B

  for (int i = 0; i < 3; ++i) {
    int base = 3 + 10 * i;
    BlockArgs a;
    a.Xb = (i == 1) ? xBb : xAb;
    a.Xf = (i == 1) ? xBf : xAf;
    a.W0t = W0t[i]; a.W1t = W1t[i]; a.W2t = W2t[i]; a.WbT = WbT[i];
    a.b0 = (const float*)d_in[base + 5];
    a.b1 = (const float*)d_in[base + 7];
    a.b2 = (const float*)d_in[base + 9];
    a.Wf = (const float*)d_in[base + 2];
    a.bfv = (const float*)d_in[base + 3];
    a.bb = (const float*)d_in[base + 1];
    a.Facc = Facc; a.addF = (i > 0);
    a.BCf = (i == 0) ? xBf : (i == 1) ? xAf : out_bc;
    a.BCb = (i == 0) ? xBb : xAb;
    a.writeBCb = (i < 2);
    fc_block_kernel<<<ROWS / 128, 256, SMEM, stream>>>(a);
  }

  fc_forecast_kernel<<<(ROWS * H_ + 255) / 256, 256, 0, stream>>>(Facc, tgfwdO,
                                                                  levelO, out_fc);
}
